// GCSAN_26697516712178
// MI455X (gfx1250) — matrix operand order repaired and checked by reading, not compiled, not run on hardware
//
#include <hip/hip_runtime.h>
#include <hip/hip_bf16.h>

// ---------------------------------------------------------------------------
// GC-SAN graph step on MI455X (gfx1250).
// One workgroup (8 wave32) per batch; LDS cut to 156KB so two workgroups
// co-reside per WGP (320KB LDS) and overlap adj HBM latency with WMMA work.
// Weights are pre-converted to f16 in global scratch by a prep kernel and
// streamed as WMMA B-operands straight from L2.
// ---------------------------------------------------------------------------

typedef __attribute__((ext_vector_type(16))) _Float16 v16h;
typedef __attribute__((ext_vector_type(8)))  _Float16 v8h;
typedef __attribute__((ext_vector_type(8)))  float    v8f;

// Problem constants
static constexpr int S_  = 128;   // sequence / nodes
static constexpr int Dm_ = 64;    // hidden dim
static constexpr int B_  = 1024;  // batch

// LDS strides (halfs / floats): rows 16B aligned, bank-rotating
static constexpr int SH64  = 72;   // [x][64] f16 (144B rows)
static constexpr int SH128 = 136;  // [x][128] f16 (272B rows)
static constexpr int SF    = 68;   // [128][64] f32 h_new (272B rows)

// LDS layout (bytes) — total 156KB => 2 workgroups per 320KB WGP
static constexpr size_t OFF_HH    = 0;                    // h f16 [128][72]
static constexpr size_t SZ_HH     = S_ * SH64 * 2;        // 18432
static constexpr size_t OFF_A1    = OFF_HH + SZ_HH;       // A1 f16 [128][136]; later h_new f32 [128][68]
static constexpr size_t SZ_A      = S_ * SH128 * 2;       // 34816 (== 128*68*4)
static constexpr size_t OFF_A2    = OFF_A1 + SZ_A;
static constexpr size_t OFF_HINT  = OFF_A2 + SZ_A;        // Hin^T f16 [64][136] ([N][K])
static constexpr size_t SZ_HT     = 64 * SH128 * 2;       // 17408
static constexpr size_t OFF_HOUTT = OFF_HINT + SZ_HT;
static constexpr size_t OFF_U0    = OFF_HOUTT + SZ_HT;    // Abin f16 [128][136] -> later inputs [Min|Mout]
static constexpr size_t OFF_RCPC  = OFF_U0 + SZ_A;        // 1/colSum f32[128]
static constexpr size_t OFF_RCPR  = OFF_RCPC + 512;       // 1/rowSum f32[128]
static constexpr size_t OFF_RED   = OFF_RCPR + 512;       // reduction scratch f32[256]
static constexpr size_t SMEM_TOTAL = OFF_RED + 1024;      // 159744 B = 156 KB
static_assert(SMEM_TOTAL <= 160u * 1024u, "need <=160KB for 2 WGs/WGP");
static_assert(SZ_A == (size_t)S_ * SF * 4, "hNew alias must fit A1 exactly");

// Global f16 weight scratch layout (elements) inside d_ws
static constexpr size_t WS_WIN  = 0;                      // W_in  [64][64]  ([N][K])
static constexpr size_t WS_WOUT = WS_WIN  + 64 * 64;      // W_out [64][64]
static constexpr size_t WS_WIH  = WS_WOUT + 64 * 64;      // w_ih  [192][128]
static constexpr size_t WS_WHH  = WS_WIH  + 192 * 128;    // w_hh  [192][64]
static constexpr size_t WS_ELEMS = WS_WHH + 192 * 64;     // 45056 halfs = 88KB

__device__ __forceinline__ v16h cat8(v8h lo, v8h hi) {
  return __builtin_shufflevector(lo, hi, 0,1,2,3,4,5,6,7,8,9,10,11,12,13,14,15);
}

// A-operand (16x32 f16) from a row-major [M][K] matrix (LDS).
// Lane: m = m0 + (lane&15), sel = lane>>4. Runs: k0+8*sel, k0+16+8*sel.
__device__ __forceinline__ v16h load_a(const _Float16* base, int strideH,
                                       int m0, int k0, int lane) {
  const int m = m0 + (lane & 15);
  const int sel = lane >> 4;
  const _Float16* p = base + m * strideH + k0;
  v8h lo = *(const v8h*)(p + sel * 8);
  v8h hi = *(const v8h*)(p + 16 + sel * 8);
  return cat8(lo, hi);
}

// B-operand (32x16 f16) from a transposed [N][K] matrix (LDS or global/L2).
// Lane: n = n0 + (lane&15), ksel = lane>>4. Runs: k0+8*ksel, k0+16+8*ksel.
__device__ __forceinline__ v16h load_b(const _Float16* baseT, int strideH,
                                       int n0, int k0, int lane) {
  const int n = n0 + (lane & 15);
  const int ksel = lane >> 4;
  const _Float16* p = baseT + n * strideH + k0;
  v8h lo = *(const v8h*)(p + ksel * 8);
  v8h hi = *(const v8h*)(p + 16 + ksel * 8);
  return cat8(lo, hi);
}

__device__ __forceinline__ v8f wmma16(v16h a, v16h b, v8f c) {
  // (neg_a, A, neg_b, B, c_mod, C, reuse_a, reuse_b)
  return __builtin_amdgcn_wmma_f32_16x16x32_f16(false, a, false, b,
                                                (short)0, c, false, false);
}

__device__ __forceinline__ float sigm(float x) {
  return 1.0f / (1.0f + __expf(-x));
}
__device__ __forceinline__ float tanh_fast(float x) {
  x = fminf(fmaxf(x, -15.0f), 15.0f);
  float e = __expf(-2.0f * x);
  return (1.0f - e) / (1.0f + e);
}

// ---------------------------------------------------------------------------
// Prep: convert all weights to f16 into global scratch (L2-resident, 88KB).
// ---------------------------------------------------------------------------
__global__ __launch_bounds__(256)
void GCSAN_prep_weights(const float* __restrict__ W_in,
                        const float* __restrict__ W_out,
                        const float* __restrict__ w_ih,
                        const float* __restrict__ w_hh,
                        _Float16* __restrict__ ws) {
  const int i = blockIdx.x * 256 + threadIdx.x;
  if (i < 4096)        ws[WS_WIN  + i]          = (_Float16)W_in [i];
  else if (i < 8192)   ws[WS_WOUT + (i - 4096)] = (_Float16)W_out[i - 4096];
  else if (i < 32768)  ws[WS_WIH  + (i - 8192)] = (_Float16)w_ih [i - 8192];
  else if (i < 45056)  ws[WS_WHH  + (i - 32768)] = (_Float16)w_hh[i - 32768];
}

// ---------------------------------------------------------------------------
// Main kernel: one block per batch.
// ---------------------------------------------------------------------------
__global__ __launch_bounds__(256)
void GCSAN_gfx1250_kernel(const int* __restrict__ adj,     // [B,128,128]
                          const int* __restrict__ items,   // [B,128]
                          const int* __restrict__ alias_,  // [B,128]
                          const float* __restrict__ E,     // [N,64]
                          const float* __restrict__ b_in,  // [64]
                          const float* __restrict__ b_out, // [64]
                          const float* __restrict__ b_ih,  // [192]
                          const float* __restrict__ b_hh,  // [192]
                          const _Float16* __restrict__ ws, // f16 weights
                          float* __restrict__ out)         // [B,128,64]
{
  extern __shared__ __align__(16) char smem[];
  _Float16* hH    = (_Float16*)(smem + OFF_HH);
  _Float16* A1    = (_Float16*)(smem + OFF_A1);
  _Float16* A2    = (_Float16*)(smem + OFF_A2);
  _Float16* HinT  = (_Float16*)(smem + OFF_HINT);
  _Float16* HoutT = (_Float16*)(smem + OFF_HOUTT);
  _Float16* Abin  = (_Float16*)(smem + OFF_U0);   // dead after A1/A2 built
  _Float16* inputs= (_Float16*)(smem + OFF_U0);   // [Min|Mout], aliases Abin
  float*    colRcp= (float*)(smem + OFF_RCPC);
  float*    rowRcp= (float*)(smem + OFF_RCPR);
  float*    red   = (float*)(smem + OFF_RED);
  float*    hNew  = (float*)(smem + OFF_A1);      // aliases A1 (dead after GEMM3)

  const _Float16* wInG  = ws + WS_WIN;   // [64][64]   ([N][K])
  const _Float16* wOutG = ws + WS_WOUT;  // [64][64]
  const _Float16* wIhG  = ws + WS_WIH;   // [192][128]
  const _Float16* wHhG  = ws + WS_WHH;   // [192][64]

  const int t    = threadIdx.x;
  const int lane = t & 31;
  const int w    = t >> 5;          // 8 waves
  const int b    = blockIdx.x;

  // ---- Phase 0: prefetch this block's 64KB adj slab (global_prefetch_b8) ----
  {
    const char* pf = (const char*)(adj + (size_t)b * S_ * S_) + t * 256;
    __builtin_prefetch(pf, 0, 3);
    __builtin_prefetch(pf + 128, 0, 3);
  }

  // ---- Phase 1: gather h = E[items[b]] -> LDS f16 (E is 25.6MB, L2-resident) ----
  {
    const int r  = t >> 1;
    const int c0 = (t & 1) * 32;
    const int it = items[b * S_ + r];
    const float4* src = (const float4*)(E + (size_t)it * Dm_ + c0);
    _Float16* dst = hH + r * SH64 + c0;
    for (int c = 0; c < 8; ++c) {
      float4 v = src[c];
      dst[4 * c + 0] = (_Float16)v.x; dst[4 * c + 1] = (_Float16)v.y;
      dst[4 * c + 2] = (_Float16)v.z; dst[4 * c + 3] = (_Float16)v.w;
    }
  }

  // ---- Phase 2a: adj -> Abin bits + row-sum partials (64MB stream: roofline) ----
  {
    const int r  = t >> 1;
    const int j0 = (t & 1) * 64;
    const int4* arow = (const int4*)(adj + ((size_t)b * S_ + r) * S_ + j0);
    float s = 0.0f;
    _Float16* drow = Abin + r * SH128 + j0;
    for (int j4 = 0; j4 < 16; ++j4) {
      int4 v = arow[j4];
      float b0 = (v.x == 2 || v.x == 4) ? 1.0f : 0.0f;
      float b1 = (v.y == 2 || v.y == 4) ? 1.0f : 0.0f;
      float b2 = (v.z == 2 || v.z == 4) ? 1.0f : 0.0f;
      float b3 = (v.w == 2 || v.w == 4) ? 1.0f : 0.0f;
      drow[4 * j4 + 0] = (_Float16)b0; drow[4 * j4 + 1] = (_Float16)b1;
      drow[4 * j4 + 2] = (_Float16)b2; drow[4 * j4 + 3] = (_Float16)b3;
      s += b0 + b1 + b2 + b3;
    }
    red[t] = s;
  }
  __syncthreads();
  if (t < S_) rowRcp[t] = 1.0f / fmaxf(red[2 * t] + red[2 * t + 1], 1.0f);
  __syncthreads();
  // ---- Phase 2b: column sums (bank-rotated stride) ----
  {
    const int c  = t >> 1;
    const int i0 = (t & 1) * 64;
    float s = 0.0f;
    for (int i = 0; i < 64; ++i) s += (float)Abin[(i0 + i) * SH128 + c];
    red[t] = s;
  }
  __syncthreads();
  if (t < S_) colRcp[t] = 1.0f / fmaxf(red[2 * t] + red[2 * t + 1], 1.0f);
  __syncthreads();

  // ---- Phase 3: A1[i][j] = Abin[j][i]*colRcp[i]; A2[i][j] = Abin[i][j]*rowRcp[j] ----
  for (int idx = t; idx < S_ * S_; idx += 256) {
    int i = idx >> 7, j = idx & 127;
    float aJI = (float)Abin[j * SH128 + i];
    float aIJ = (float)Abin[i * SH128 + j];
    A1[i * SH128 + j] = (_Float16)(aJI * colRcp[i]);
    A2[i * SH128 + j] = (_Float16)(aIJ * rowRcp[j]);
  }
  __syncthreads();

  // ---- Phase 4: Hin = h*W_in^T + b_in, Hout = h*W_out^T + b_out; store transposed.
  //      B-operands come straight from global f16 weights (L2/WGP$ broadcast). ----
  {
    const int m0  = w * 16;
    const int sel = lane >> 4;
    for (int nt = 0; nt < 4; ++nt) {
      const int n0 = nt * 16;
      v8f ci = {}, co = {};
      for (int k0 = 0; k0 < Dm_; k0 += 32) {
        v16h a = load_a(hH, SH64, m0, k0, lane);
        ci = wmma16(a, load_b(wInG,  Dm_, n0, k0, lane), ci);
        co = wmma16(a, load_b(wOutG, Dm_, n0, k0, lane), co);
      }
      const int n = n0 + (lane & 15);
      const float bi = b_in[n], bo = b_out[n];
      v8h pi, po;
      for (int v = 0; v < 8; ++v) {
        pi[v] = (_Float16)(ci[v] + bi);
        po[v] = (_Float16)(co[v] + bo);
      }
      *(v8h*)(HinT  + n * SH128 + m0 + 8 * sel) = pi;   // [N][K] for next B-op
      *(v8h*)(HoutT + n * SH128 + m0 + 8 * sel) = po;
    }
  }
  __syncthreads();

  // ---- Phase 5: Min = A1*Hin, Mout = A2*Hout -> inputs = [Min|Mout] row-major ----
  {
    const int m0  = w * 16;
    const int sel = lane >> 4;
    for (int nt = 0; nt < 4; ++nt) {
      const int n0 = nt * 16;
      v8f ci = {}, co = {};
      for (int k0 = 0; k0 < S_; k0 += 32) {
        ci = wmma16(load_a(A1, SH128, m0, k0, lane),
                    load_b(HinT,  SH128, n0, k0, lane), ci);
        co = wmma16(load_a(A2, SH128, m0, k0, lane),
                    load_b(HoutT, SH128, n0, k0, lane), co);
      }
      const int n = n0 + (lane & 15);
      for (int v = 0; v < 8; ++v) {
        const int m = m0 + v + 8 * sel;
        inputs[m * SH128 + n]      = (_Float16)ci[v];
        inputs[m * SH128 + 64 + n] = (_Float16)co[v];
      }
    }
  }
  __syncthreads();

  // ---- Phase 6: Gi = inputs*w_ih^T, Gh = h*w_hh^T, fused GRU -> hNew (f32) ----
  {
    const int m0  = w * 16;
    const int sel = lane >> 4;
    for (int nt = 0; nt < 4; ++nt) {
      const int n0 = nt * 16;
      v8f gr = {}, gu = {}, gn = {};
      for (int k0 = 0; k0 < 2 * Dm_; k0 += 32) {
        v16h a = load_a(inputs, SH128, m0, k0, lane);
        gr = wmma16(a, load_b(wIhG, 2 * Dm_, n0,       k0, lane), gr);
        gu = wmma16(a, load_b(wIhG, 2 * Dm_, 64 + n0,  k0, lane), gu);
        gn = wmma16(a, load_b(wIhG, 2 * Dm_, 128 + n0, k0, lane), gn);
      }
      v8f hr = {}, hu = {}, hn = {};
      for (int k0 = 0; k0 < Dm_; k0 += 32) {
        v16h a = load_a(hH, SH64, m0, k0, lane);
        hr = wmma16(a, load_b(wHhG, Dm_, n0,       k0, lane), hr);
        hu = wmma16(a, load_b(wHhG, Dm_, 64 + n0,  k0, lane), hu);
        hn = wmma16(a, load_b(wHhG, Dm_, 128 + n0, k0, lane), hn);
      }
      const int n = n0 + (lane & 15);
      const float bir = b_ih[n], biu = b_ih[64 + n], bin_ = b_ih[128 + n];
      const float bhr = b_hh[n], bhu = b_hh[64 + n], bhn  = b_hh[128 + n];
      for (int v = 0; v < 8; ++v) {
        const int m = m0 + v + 8 * sel;
        float r  = sigm(gr[v] + bir + hr[v] + bhr);
        float u  = sigm(gu[v] + biu + hu[v] + bhu);
        float nn = tanh_fast(gn[v] + bin_ + r * (hn[v] + bhn));
        float h0 = (float)hH[m * SH64 + n];
        hNew[m * SF + n] = (1.0f - u) * h0 + u * nn;
      }
    }
  }
  __syncthreads();

  // ---- Phase 7: out[b,i,:] = hNew[alias[b,i],:] ----
  {
    const int i  = t >> 1;
    const int c0 = (t & 1) * 32;
    const int a  = alias_[b * S_ + i];
    const float* src = hNew + a * SF + c0;
    float* dst = out + ((size_t)b * S_ + i) * Dm_ + c0;
    for (int c = 0; c < 32; c += 4)
      *(float4*)(dst + c) = *(const float4*)(src + c);
  }
}

extern "C" void kernel_launch(void* const* d_in, const int* in_sizes, int n_in,
                              void* d_out, int out_size, void* d_ws, size_t ws_size,
                              hipStream_t stream) {
  (void)in_sizes; (void)n_in; (void)out_size; (void)ws_size;
  const int*   adj    = (const int*)  d_in[0];
  const int*   items  = (const int*)  d_in[1];
  const int*   alias_ = (const int*)  d_in[2];
  const float* E      = (const float*)d_in[3];
  const float* W_in   = (const float*)d_in[4];
  const float* b_in   = (const float*)d_in[5];
  const float* W_out  = (const float*)d_in[6];
  const float* b_out  = (const float*)d_in[7];
  const float* w_ih   = (const float*)d_in[8];
  const float* b_ih   = (const float*)d_in[9];
  const float* w_hh   = (const float*)d_in[10];
  const float* b_hh   = (const float*)d_in[11];
  float* out = (float*)d_out;
  _Float16* ws = (_Float16*)d_ws;   // needs 88KB; d_ws is preallocated scratch

  // Stage 1: weights -> f16 scratch (stream-ordered before main kernel).
  GCSAN_prep_weights<<<dim3((WS_ELEMS + 255) / 256), dim3(256), 0, stream>>>(
      W_in, W_out, w_ih, w_hh, ws);

  // Stage 2: fused graph step, one block per batch.
  GCSAN_gfx1250_kernel<<<dim3(B_), dim3(256), SMEM_TOTAL, stream>>>(
      adj, items, alias_, E, b_in, b_out, b_ih, b_hh, ws, out);
}
